// GNN_w_Dense_58703613002017
// MI455X (gfx1250) — matrix operand order repaired and checked by reading, not compiled, not run on hardware
//
#include <hip/hip_runtime.h>
#include <hip/hip_bf16.h>

typedef __attribute__((ext_vector_type(16))) _Float16 v16h;
typedef __attribute__((ext_vector_type(8)))  _Float16 v8h;
typedef __attribute__((ext_vector_type(8)))  float    v8f;

union V16H { v16h v; v8h h[2]; };

__device__ __forceinline__ void atomAddF(float* p, float v) {
  __hip_atomic_fetch_add(p, v, __ATOMIC_RELAXED, __HIP_MEMORY_SCOPE_AGENT);
}

// ---------------------------------------------------------------------------
// Generic f32 -> f16 convert
// ---------------------------------------------------------------------------
__global__ void k_f32_to_f16(const float* __restrict__ in, _Float16* __restrict__ out,
                             long long n) {
  long long i = (long long)blockIdx.x * blockDim.x + threadIdx.x;
  if (i < n) out[i] = (_Float16)in[i];
}

// ---------------------------------------------------------------------------
// Count occurrences of idx[i] (edge in-degree / graph sizes)
// ---------------------------------------------------------------------------
__global__ void k_count(const int* __restrict__ idx, float* __restrict__ cnt, long long n) {
  long long i = (long long)blockIdx.x * blockDim.x + threadIdx.x;
  if (i < n) atomAddF(&cnt[idx[i]], 1.0f);
}

// ---------------------------------------------------------------------------
// Edge scatter: msg[dst[e]][f] += feat[src[e]][f], F = 128
// ---------------------------------------------------------------------------
__global__ void k_scatter(const int* __restrict__ src, const int* __restrict__ dst,
                          const float* __restrict__ feat, float* __restrict__ msg,
                          long long total) {
  long long i = (long long)blockIdx.x * blockDim.x + threadIdx.x;
  if (i >= total) return;
  long long e = i >> 7;
  int f = (int)(i & 127);
  float v = feat[(long long)src[e] * 128 + f];
  atomAddF(&msg[(long long)dst[e] * 128 + f], v);
}

// ---------------------------------------------------------------------------
// Mean + convert to f16:  out[n][f] = (f16)(msg[n][f] / max(cnt[n],1))
// ---------------------------------------------------------------------------
__global__ void k_mean_to_h(const float* __restrict__ msg, const float* __restrict__ cnt,
                            _Float16* __restrict__ out, long long total) {
  long long i = (long long)blockIdx.x * blockDim.x + threadIdx.x;
  if (i >= total) return;
  long long node = i >> 7;
  float s = msg[i] / fmaxf(cnt[node], 1.0f);
  out[i] = (_Float16)s;
}

// ---------------------------------------------------------------------------
// WMMA GEMM:  out[m][n] = sum_k A1[m][k]*W1[n][k]  (+ sum_k A2[m][k]*W2[n][k]) + bias[n]
// A: [Mpad x 128] f16 row-major; W: [Hout x 128] f16 row-major; K = 128 fixed.
// One wave computes a 16x64 output strip (4 accumulators) so each A fragment
// is loaded once and reused across 4 N-tiles.  EXEC is all-1s at every WMMA.
// ---------------------------------------------------------------------------
__global__ __launch_bounds__(256) void k_wmma_gemm(
    const _Float16* __restrict__ A1, const _Float16* __restrict__ W1,
    const _Float16* __restrict__ A2, const _Float16* __restrict__ W2,
    const float* __restrict__ bias, float* __restrict__ out,
    int M, int Hout) {
  const int lane = threadIdx.x & 31;
  const int wave = threadIdx.x >> 5;
  const int tn4_count = Hout >> 6;                 // 64-wide column groups
  const int tm_count  = (M + 15) >> 4;
  const int tile = blockIdx.x * 8 + wave;
  if (tile >= tm_count * tn4_count) return;        // wave-uniform
  const int tm  = tile / tn4_count;
  const int tn4 = tile - tm * tn4_count;
  const int hi = lane >> 4;                        // 0 | 1
  const int l  = lane & 15;

  const _Float16* a1 = A1 + (size_t)(tm * 16 + l) * 128;
  // base weight row for this lane within the 64-wide strip
  const _Float16* w1 = W1 + (size_t)(tn4 * 64 + l) * 128;

  v8f c[4] = {{}, {}, {}, {}};

#pragma unroll
  for (int k0 = 0; k0 < 128; k0 += 32) {
    V16H a;
    // A 16x32: lane holds row l, K = k0 + 16*hi + 0..15
    a.h[0] = *(const v8h*)(a1 + k0 + hi * 16);
    a.h[1] = *(const v8h*)(a1 + k0 + hi * 16 + 8);
#pragma unroll
    for (int j = 0; j < 4; ++j) {
      V16H b;
      // B 32x16: lane holds column l of tile j, K = k0 + 16*hi + 0..15
      const _Float16* wj = w1 + (size_t)(j * 16) * 128;
      b.h[0] = *(const v8h*)(wj + k0 + hi * 16);
      b.h[1] = *(const v8h*)(wj + k0 + hi * 16 + 8);
      c[j] = __builtin_amdgcn_wmma_f32_16x16x32_f16(false, a.v, false, b.v,
                                                    (short)0, c[j], false, false);
    }
  }
  if (A2 != nullptr) {
    const _Float16* a2 = A2 + (size_t)(tm * 16 + l) * 128;
    const _Float16* w2 = W2 + (size_t)(tn4 * 64 + l) * 128;
#pragma unroll
    for (int k0 = 0; k0 < 128; k0 += 32) {
      V16H a;
      a.h[0] = *(const v8h*)(a2 + k0 + hi * 16);
      a.h[1] = *(const v8h*)(a2 + k0 + hi * 16 + 8);
#pragma unroll
      for (int j = 0; j < 4; ++j) {
        V16H b;
        const _Float16* wj = w2 + (size_t)(j * 16) * 128;
        b.h[0] = *(const v8h*)(wj + k0 + hi * 16);
        b.h[1] = *(const v8h*)(wj + k0 + hi * 16 + 8);
        c[j] = __builtin_amdgcn_wmma_f32_16x16x32_f16(false, a.v, false, b.v,
                                                      (short)0, c[j], false, false);
      }
    }
  }

#pragma unroll
  for (int j = 0; j < 4; ++j) {
    const int ncol = tn4 * 64 + j * 16 + l;
    const float bn = bias[ncol];
#pragma unroll
    for (int r = 0; r < 8; ++r) {
      int m = tm * 16 + hi * 8 + r;                // C layout: VGPR r -> M = r + 8*hi
      if (m < M) out[(size_t)m * Hout + ncol] = c[j][r] + bn;
    }
  }
}

// ---------------------------------------------------------------------------
// Row-wise L2 normalize (in place), optional ReLU.  One wave per row.
// ---------------------------------------------------------------------------
__global__ void k_rownorm(float* __restrict__ x, int M, int F, int relu) {
  int row = blockIdx.x * 8 + (threadIdx.x >> 5);
  if (row >= M) return;
  int lane = threadIdx.x & 31;
  float* p = x + (size_t)row * F;
  float s = 0.f;
  for (int j = lane; j < F; j += 32) { float v = p[j]; s += v * v; }
  for (int o = 16; o > 0; o >>= 1) s += __shfl_xor(s, o, 32);
  float scale = 1.0f / fmaxf(sqrtf(s), 1e-12f);
  for (int j = lane; j < F; j += 32) {
    float v = p[j] * scale;
    if (relu) v = fmaxf(v, 0.f);
    p[j] = v;
  }
}

// ---------------------------------------------------------------------------
// Global pool scatter:  pooled[batch[n]][c] += h[n][c], C = 256
// ---------------------------------------------------------------------------
__global__ void k_pool(const int* __restrict__ batch, const float* __restrict__ h,
                       float* __restrict__ pooled, long long total) {
  long long i = (long long)blockIdx.x * blockDim.x + threadIdx.x;
  if (i >= total) return;
  long long n = i >> 8;
  int c = (int)(i & 255);
  atomAddF(&pooled[(long long)batch[n] * 256 + c], h[i]);
}

// ---------------------------------------------------------------------------
// Head:  out[g] = sigmoid( (pooled[g]/max(gcnt,1)) . linW + linb )
// ---------------------------------------------------------------------------
__global__ void k_head(const float* __restrict__ pooled, const float* __restrict__ gcnt,
                       const float* __restrict__ W, const float* __restrict__ b,
                       float* __restrict__ out, int G) {
  int g = blockIdx.x * 8 + (threadIdx.x >> 5);
  if (g >= G) return;
  int lane = threadIdx.x & 31;
  float inv = 1.0f / fmaxf(gcnt[g], 1.0f);
  float s = 0.f;
  for (int j = lane; j < 256; j += 32) s += pooled[(long long)g * 256 + j] * W[j];
  for (int o = 16; o > 0; o >>= 1) s += __shfl_xor(s, o, 32);
  if (lane == 0) {
    float z = s * inv + b[0];
    out[g] = 1.0f / (1.0f + expf(-z));
  }
}

// ---------------------------------------------------------------------------
extern "C" void kernel_launch(void* const* d_in, const int* in_sizes, int n_in,
                              void* d_out, int out_size, void* d_ws, size_t ws_size,
                              hipStream_t stream) {
  const float* x      = (const float*)d_in[0];
  const int*   eidx   = (const int*)d_in[1];
  const int*   batch  = (const int*)d_in[2];
  const float* lin1_W = (const float*)d_in[3];
  const float* lin1_b = (const float*)d_in[4];
  const float* c1_Wl  = (const float*)d_in[5];
  const float* c1_bl  = (const float*)d_in[6];
  const float* c1_Wr  = (const float*)d_in[7];
  const float* c2_Wl  = (const float*)d_in[8];
  const float* c2_bl  = (const float*)d_in[9];
  const float* c2_Wr  = (const float*)d_in[10];
  const float* lin_W  = (const float*)d_in[11];
  const float* lin_b  = (const float*)d_in[12];
  float* out = (float*)d_out;

  const int H = 128;
  const long long N = in_sizes[0] / H;
  const long long E = in_sizes[1] / 2;
  const int G = out_size;
  const long long Npad = (N + 15) & ~15LL;
  const int* src = eidx;
  const int* dst = eidx + E;

  // -------- workspace carve-up (256B aligned) --------
  size_t off = 0;
  auto carve = [&](size_t bytes) -> void* {
    off = (off + 255) & ~(size_t)255;
    void* p = (char*)d_ws + off;
    off += bytes;
    return p;
  };
  float*    h_f32  = (float*)carve((size_t)Npad * 128 * 4);   // h1, then h2 (f32)
  float*    msg    = (float*)carve((size_t)Npad * 128 * 4);   // scatter accumulator
  float*    cnt    = (float*)carve((size_t)Npad * 4);         // in-degree
  float*    h3     = (float*)carve((size_t)Npad * 256 * 4);   // conv2 output
  float*    pooled = (float*)carve((size_t)G * 256 * 4);
  float*    gcnt   = (float*)carve((size_t)G * 4);
  _Float16* h_h    = (_Float16*)carve((size_t)Npad * 128 * 2); // x/h1/h2 in f16
  _Float16* aggr_h = (_Float16*)carve((size_t)Npad * 128 * 2); // mean-aggregated f16
  _Float16* wb     = (_Float16*)carve((size_t)(3 * 16384 + 2 * 32768) * 2);
  _Float16* w_lin1 = wb;
  _Float16* w_c1l  = wb + 16384;
  _Float16* w_c1r  = wb + 2 * 16384;
  _Float16* w_c2l  = wb + 3 * 16384;
  _Float16* w_c2r  = wb + 3 * 16384 + 32768;
  (void)ws_size; (void)n_in;

  const int TB = 256;
  auto blk1 = [&](long long n) { return (unsigned)((n + TB - 1) / TB); };
  auto gemm_blocks = [&](long long M, int Hout) {
    long long tiles = ((M + 15) / 16) * (Hout / 64);   // 16x64 strip per wave
    return (unsigned)((tiles + 7) / 8);
  };

  // -------- weights -> f16 --------
  k_f32_to_f16<<<blk1(16384), TB, 0, stream>>>(lin1_W, w_lin1, 16384);
  k_f32_to_f16<<<blk1(16384), TB, 0, stream>>>(c1_Wl, w_c1l, 16384);
  k_f32_to_f16<<<blk1(16384), TB, 0, stream>>>(c1_Wr, w_c1r, 16384);
  k_f32_to_f16<<<blk1(32768), TB, 0, stream>>>(c2_Wl, w_c2l, 32768);
  k_f32_to_f16<<<blk1(32768), TB, 0, stream>>>(c2_Wr, w_c2r, 32768);

  // -------- in-degree (used by both convs) --------
  hipMemsetAsync(cnt, 0, (size_t)Npad * 4, stream);
  k_count<<<blk1(E), TB, 0, stream>>>(dst, cnt, E);

  // -------- lin1: h1 = x @ lin1_W.T + b --------
  k_f32_to_f16<<<blk1(N * 128), TB, 0, stream>>>(x, h_h, N * 128);
  k_wmma_gemm<<<gemm_blocks(N, 128), TB, 0, stream>>>(
      h_h, w_lin1, nullptr, nullptr, lin1_b, h_f32, (int)N, 128);

  // -------- conv1 --------
  hipMemsetAsync(msg, 0, (size_t)Npad * 128 * 4, stream);
  k_scatter<<<blk1(E * 128), TB, 0, stream>>>(src, dst, h_f32, msg, E * 128);
  k_mean_to_h<<<blk1(N * 128), TB, 0, stream>>>(msg, cnt, aggr_h, N * 128);
  k_f32_to_f16<<<blk1(N * 128), TB, 0, stream>>>(h_f32, h_h, N * 128);
  k_wmma_gemm<<<gemm_blocks(N, 128), TB, 0, stream>>>(
      aggr_h, w_c1l, h_h, w_c1r, c1_bl, h_f32, (int)N, 128);
  k_rownorm<<<(unsigned)((N + 7) / 8), TB, 0, stream>>>(h_f32, (int)N, 128, /*relu=*/1);

  // -------- conv2 --------
  hipMemsetAsync(msg, 0, (size_t)Npad * 128 * 4, stream);
  k_scatter<<<blk1(E * 128), TB, 0, stream>>>(src, dst, h_f32, msg, E * 128);
  k_mean_to_h<<<blk1(N * 128), TB, 0, stream>>>(msg, cnt, aggr_h, N * 128);
  k_f32_to_f16<<<blk1(N * 128), TB, 0, stream>>>(h_f32, h_h, N * 128);
  k_wmma_gemm<<<gemm_blocks(N, 256), TB, 0, stream>>>(
      aggr_h, w_c2l, h_h, w_c2r, c2_bl, h3, (int)N, 256);
  k_rownorm<<<(unsigned)((N + 7) / 8), TB, 0, stream>>>(h3, (int)N, 256, /*relu=*/0);

  // -------- global mean pool + head --------
  hipMemsetAsync(pooled, 0, (size_t)G * 256 * 4, stream);
  hipMemsetAsync(gcnt, 0, (size_t)G * 4, stream);
  k_count<<<blk1(N), TB, 0, stream>>>(batch, gcnt, N);
  k_pool<<<blk1(N * 256), TB, 0, stream>>>(batch, h3, pooled, N * 256);
  k_head<<<(unsigned)((G + 7) / 8), TB, 0, stream>>>(pooled, gcnt, lin_W, lin_b, out, G);
}
